// MajFC_463856468334
// MI455X (gfx1250) — 8-bit matrix operand order repaired and checked by reading, compiled, not run on hardware
//
#include <hip/hip_runtime.h>

typedef int v8i __attribute__((ext_vector_type(8)));

#define C_IN   3072
#define C_OUT  1024
#define BATCH  256
#define GRP    1024
#define KTOT   4096

__device__ __forceinline__ unsigned int sign_byte(unsigned int signbit) {
  return signbit ? 0xFFu : 0x01u;
}

__global__ __launch_bounds__(256) void pack_signs(const float* __restrict__ src,
                                                  signed char* __restrict__ dst,
                                                  int nrows, int neg_prod) {
  int tid = blockIdx.x * 256 + threadIdx.x;
  int row = tid >> 10;
  int q   = tid & 1023;
  int j   = q << 2;
  if (row >= nrows) return;

  const float* xr = src + (size_t)row * C_IN;
  unsigned int bytes;

  if (j < C_IN) {
    float4 v = *(const float4*)(xr + j);
    unsigned int s0 = __float_as_uint(v.x) >> 31;
    unsigned int s1 = __float_as_uint(v.y) >> 31;
    unsigned int s2 = __float_as_uint(v.z) >> 31;
    unsigned int s3 = __float_as_uint(v.w) >> 31;
    bytes = sign_byte(s0) | (sign_byte(s1) << 8) |
            (sign_byte(s2) << 16) | (sign_byte(s3) << 24);
  } else {
    int g = j - C_IN;
    const float* p = xr + 3 * g;
    float4 v0 = *(const float4*)(p);
    float4 v1 = *(const float4*)(p + 4);
    float4 v2 = *(const float4*)(p + 8);
    unsigned int nf = neg_prod ? 0x80000000u : 0u;
    unsigned int t0 = (__float_as_uint(v0.x) ^ __float_as_uint(v0.y) ^ __float_as_uint(v0.z) ^ nf) >> 31;
    unsigned int t1 = (__float_as_uint(v0.w) ^ __float_as_uint(v1.x) ^ __float_as_uint(v1.y) ^ nf) >> 31;
    unsigned int t2 = (__float_as_uint(v1.z) ^ __float_as_uint(v1.w) ^ __float_as_uint(v2.x) ^ nf) >> 31;
    unsigned int t3 = (__float_as_uint(v2.y) ^ __float_as_uint(v2.z) ^ __float_as_uint(v2.w) ^ nf) >> 31;
    bytes = sign_byte(t0) | (sign_byte(t1) << 8) |
            (sign_byte(t2) << 16) | (sign_byte(t3) << 24);
  }
  *(unsigned int*)(dst + (size_t)row * KTOT + j) = bytes;
}

__global__ __launch_bounds__(256) void maj_wmma_gemm(const signed char* __restrict__ A,
                                                     const signed char* __restrict__ B,
                                                     float* __restrict__ out) {
  const int lane = threadIdx.x & 31;
  const int wave = threadIdx.x >> 5;
  const int tile = blockIdx.x * 8 + wave;
  const int tm = (tile & 15) << 4;
  const int tn = (tile >> 4) << 5;

  const int m  = lane & 15;
  const int kh = lane >> 4;

  const signed char* arow  = A + (size_t)(tm + m) * KTOT + kh * 16;
  const signed char* b0row = B + (size_t)(tn + m) * KTOT + kh * 16;
  const signed char* b1row = b0row + (size_t)16 * KTOT;

  v8i acc0 = {0, 0, 0, 0, 0, 0, 0, 0};
  v8i acc1 = {0, 0, 0, 0, 0, 0, 0, 0};

  for (int kk = 0; kk < KTOT; kk += 64) {
    const uint2* pa = (const uint2*)(arow + kk);
    uint2 a01 = pa[0];
    uint2 a23 = pa[1];
    uint2 a45 = pa[4];
    uint2 a67 = pa[5];
    v8i av = {(int)a01.x, (int)a01.y, (int)a23.x, (int)a23.y,
              (int)a45.x, (int)a45.y, (int)a67.x, (int)a67.y};

    const int4* pb0 = (const int4*)(b0row + kk);
    int4 b0l = pb0[0];
    int4 b0h = pb0[2];
    v8i bv0 = {b0l.x, b0l.y, b0l.z, b0l.w, b0h.x, b0h.y, b0h.z, b0h.w};

    const int4* pb1 = (const int4*)(b1row + kk);
    int4 b1l = pb1[0];
    int4 b1h = pb1[2];
    v8i bv1 = {b1l.x, b1l.y, b1l.z, b1l.w, b1h.x, b1h.y, b1h.z, b1h.w};

    acc0 = __builtin_amdgcn_wmma_i32_16x16x64_iu8(true, av, true, bv0, acc0, false, false);
    acc1 = __builtin_amdgcn_wmma_i32_16x16x64_iu8(true, av, true, bv1, acc1, false, false);
  }

  const int n  = lane & 15;
  const int mb = kh * 8;
  float* o = out + (size_t)(tm + mb) * C_OUT + tn + n;
#pragma unroll
  for (int r = 0; r < 8; ++r) {
    o[(size_t)r * C_OUT]      = 1.125f * (float)acc0[r];
    o[(size_t)r * C_OUT + 16] = 1.125f * (float)acc1[r];
  }
}

extern "C" void kernel_launch(void* const* d_in, const int* in_sizes, int n_in,
                              void* d_out, int out_size, void* d_ws, size_t ws_size,
                              hipStream_t stream) {
  const float* x = (const float*)d_in[0];
  const float* w = (const float*)d_in[1];

  signed char* Apk = (signed char*)d_ws;
  signed char* Bpk = Apk + (size_t)BATCH * KTOT;

  pack_signs<<<(BATCH * KTOT / 4) / 256, 256, 0, stream>>>(x, Apk, BATCH, 0);
  pack_signs<<<(C_OUT * KTOT / 4) / 256, 256, 0, stream>>>(w, Bpk, C_OUT, 1);

  maj_wmma_gemm<<<64, 256, 0, stream>>>(Apk, Bpk, (float*)d_out);
}
